// ArchitectureGNN_1846835937794
// MI455X (gfx1250) — matrix operand order repaired and checked by reading, not compiled, not run on hardware
//
#include <hip/hip_runtime.h>
#include <math.h>

// ---------------- problem constants ----------------
#define NN   40000
#define EE   640000
#define HID  128
#define HEADS 4
#define GG   128
#define NODE_DIM 4
#define EPS  1e-5f

typedef __attribute__((ext_vector_type(16))) __bf16 v16bf;
typedef __attribute__((ext_vector_type(8)))  float  v8f;

// ---------------- small helpers ----------------
__device__ __forceinline__ float leaky02(float x) { return x > 0.f ? x : 0.2f * x; }

__device__ __forceinline__ void atomicMaxF(float* addr, float v) {
    if (v >= 0.f) atomicMax((int*)addr, __float_as_int(v));
    else          atomicMin((unsigned int*)addr, __float_as_uint(v));
}

// CDNA5 async global->LDS copy (16B). VDST operand is an LDS byte address in a
// VGPR (low 32 bits of a generic pointer to __shared__ memory per ISA 10.2).
__device__ __forceinline__ void async_ld16(unsigned lds_off, const void* gsrc) {
    asm volatile("global_load_async_to_lds_b128 %0, %1, off"
                 :: "v"(lds_off), "v"(gsrc) : "memory");
}
__device__ __forceinline__ void wait_async0() {
    asm volatile("s_wait_asynccnt 0x0" ::: "memory");
}

// ---------------- generic fill ----------------
__global__ void k_fill(float* p, float v, int n) {
    int i = blockIdx.x * blockDim.x + threadIdx.x;
    if (i < n) p[i] = v;
}

// ---------------- degree / norm ----------------
__global__ void k_deg_scatter(const int* __restrict__ dst, float* deg, int e) {
    int i = blockIdx.x * blockDim.x + threadIdx.x;
    if (i < e) atomicAdd(&deg[dst[i]], 1.0f);
}
__global__ void k_rsqrt_inplace(float* p, int n) {
    int i = blockIdx.x * blockDim.x + threadIdx.x;
    if (i < n) p[i] = rsqrtf(p[i]);
}

// ---------------- embed: h = x @ W(4x128) + b ----------------
__global__ void k_embed(const float* __restrict__ x, const float* __restrict__ W,
                        const float* __restrict__ b, float* __restrict__ h) {
    int i = blockIdx.x * blockDim.x + threadIdx.x;
    if (i >= NN * HID) return;
    int n = i >> 7, f = i & 127;
    float s = b[f];
#pragma unroll
    for (int k = 0; k < NODE_DIM; ++k) s += x[n * NODE_DIM + k] * W[k * HID + f];
    h[i] = s;
}

// ---------------- convert + transpose weights to bf16 (Bt[n][k] = W[k][n]) ----------------
__global__ void k_wt_bf16(const float* __restrict__ W, __bf16* __restrict__ Bt, int NC) {
    int n = blockIdx.x;           // output column, 0..NC-1
    int k = threadIdx.x;          // 0..127
    Bt[n * HID + k] = (__bf16)W[k * NC + n];
}

// ---------------- bf16 WMMA GEMM: C[M x NC] = A[M x 128] @ B[128 x NC] ----------------
// One block per 16-row M-tile. A tile staged once into LDS via async copy;
// A fragments hoisted to registers; each wave sweeps n-tiles (nt += 8).
// blockDim = 256 (8 waves); grid = M/16.
__global__ void k_gemm_bf16(const float* __restrict__ A, const __bf16* __restrict__ Bt,
                            float* __restrict__ C, int NC, int ntiles) {
    __shared__ __align__(16) float Atile[16 * HID];   // 8 KB
    int tid   = threadIdx.x;
    int lane  = tid & 31;
    int wave  = tid >> 5;
    int mbase = blockIdx.x * 16;

    // stage A tile (16x128 f32 = 8KB) global -> LDS, async, 16B per thread x2
    {
        const char* gsrc = (const char*)(A + (size_t)mbase * HID);
        unsigned lds_base = (unsigned)(size_t)&Atile[0];
#pragma unroll
        for (int i = 0; i < 2; ++i) {
            unsigned byt = (unsigned)(tid + i * 256) * 16u;
            async_ld16(lds_base + byt, gsrc + byt);
        }
        wait_async0();
        __syncthreads();
    }

    int row   = lane & 15;        // M-row within tile (A layout)
    int khalf = lane >> 4;        // K-half owned by this lane (A layout)
    int col   = lane & 15;        // N-col (B/C layout)
    int kgrp  = lane >> 4;        // K-group of 16 (B layout)

    // A fragments: lane<16 -> K {0..15}+32ks ; lane>=16 -> K {16..31}+32ks
    v16bf afrag[4];
#pragma unroll
    for (int ks = 0; ks < 4; ++ks) {
        const float* ar = &Atile[row * HID + ks * 32 + khalf * 16];
#pragma unroll
        for (int j = 0; j < 8; ++j) {
            afrag[ks][j]     = (__bf16)ar[j];
            afrag[ks][j + 8] = (__bf16)ar[j + 8];
        }
    }

    for (int nt = wave; nt < ntiles; nt += 8) {
        int nbase = nt * 16;
        v8f acc = {};
#pragma unroll
        for (int ks = 0; ks < 4; ++ks) {
            // B fragment: contiguous 16 K-values at fixed column (transposed weights)
            const __bf16* bp = Bt + (size_t)(nbase + col) * HID + ks * 32 + kgrp * 16;
            v16bf bfrag = *(const v16bf*)bp;
            acc = __builtin_amdgcn_wmma_f32_16x16x32_bf16(false, afrag[ks], false, bfrag,
                                                          (short)0, acc, false, false);
        }
        // C/D layout: VGPR r -> row r (lanes 0-15) / r+8 (lanes 16-31), col = lane&15
        float* crow = C + (size_t)(mbase + khalf * 8) * NC + nbase + col;
#pragma unroll
        for (int r = 0; r < 8; ++r) crow[(size_t)r * NC] = acc[r];
    }
}

// ---------------- GCN aggregate ----------------
__global__ void k_conv_init(const float* __restrict__ t, const float* __restrict__ dinv,
                            const float* __restrict__ bias, float* __restrict__ agg) {
    int i = blockIdx.x * blockDim.x + threadIdx.x;
    if (i >= NN * HID) return;
    int n = i >> 7, f = i & 127;
    float dn = dinv[n];
    agg[i] = bias[f] + t[i] * dn * dn;
}
__global__ void k_conv_scatter(const int* __restrict__ src, const int* __restrict__ dst,
                               const float* __restrict__ t, const float* __restrict__ dinv,
                               float* __restrict__ agg) {
    int e = blockIdx.x;
    int f = threadIdx.x;
    int s = src[e], d = dst[e];
    float nrm = dinv[s] * dinv[d];
    atomicAdd(&agg[(size_t)d * HID + f], t[(size_t)s * HID + f] * nrm);
}

// ---------------- batchnorm ----------------
__global__ void k_bn_partial(const float* __restrict__ agg, float* musum, float* sqsum) {
    int f = threadIdx.x;                     // 0..127
    int chunk = (NN + gridDim.x - 1) / gridDim.x;
    int n0 = blockIdx.x * chunk;
    int n1 = min(NN, n0 + chunk);
    float s = 0.f, q = 0.f;
    for (int n = n0; n < n1; ++n) {
        float v = agg[(size_t)n * HID + f];
        s += v; q += v * v;
    }
    atomicAdd(&musum[f], s);
    atomicAdd(&sqsum[f], q);
}
__global__ void k_bn_relu(const float* __restrict__ agg, const float* __restrict__ musum,
                          const float* __restrict__ sqsum, const float* __restrict__ gamma,
                          const float* __restrict__ beta, float* __restrict__ h) {
    int i = blockIdx.x * blockDim.x + threadIdx.x;
    if (i >= NN * HID) return;
    int f = i & 127;
    const float invN = 1.0f / (float)NN;
    float mu  = musum[f] * invN;
    float var = sqsum[f] * invN - mu * mu;
    float v = (agg[i] - mu) * rsqrtf(var + EPS) * gamma[f] + beta[f];
    h[i] = fmaxf(v, 0.f);
}

// ---------------- GAT ----------------
__global__ void k_att_scores(const float* __restrict__ hh, const float* __restrict__ att_src,
                             const float* __restrict__ att_dst,
                             float* __restrict__ asrc, float* __restrict__ adst) {
    int i = blockIdx.x * blockDim.x + threadIdx.x;
    if (i >= NN * HEADS) return;
    int n = i >> 2, hd = i & 3;
    const float* hrow = hh + (size_t)n * (HEADS * HID) + hd * HID;
    const float* as = att_src + hd * HID;
    const float* ad = att_dst + hd * HID;
    float s = 0.f, d = 0.f;
    for (int c = 0; c < HID; ++c) { float v = hrow[c]; s += v * as[c]; d += v * ad[c]; }
    asrc[i] = s; adst[i] = d;
}
__global__ void k_gat_m_init(const float* __restrict__ asrc, const float* __restrict__ adst,
                             float* __restrict__ m) {
    int i = blockIdx.x * blockDim.x + threadIdx.x;
    if (i < NN * HEADS) m[i] = leaky02(asrc[i] + adst[i]);   // self-loop edge
}
__global__ void k_gat_max(const int* __restrict__ src, const int* __restrict__ dst,
                          const float* __restrict__ asrc, const float* __restrict__ adst,
                          float* __restrict__ m) {
    int i = blockIdx.x * blockDim.x + threadIdx.x;
    if (i >= EE) return;
    int s = src[i], d = dst[i];
#pragma unroll
    for (int hd = 0; hd < HEADS; ++hd)
        atomicMaxF(&m[d * HEADS + hd], leaky02(asrc[s * HEADS + hd] + adst[d * HEADS + hd]));
}
__global__ void k_gat_s_init(const float* __restrict__ asrc, const float* __restrict__ adst,
                             const float* __restrict__ m, float* __restrict__ ssum) {
    int i = blockIdx.x * blockDim.x + threadIdx.x;
    if (i < NN * HEADS) ssum[i] = expf(leaky02(asrc[i] + adst[i]) - m[i]);
}
__global__ void k_gat_s_scatter(const int* __restrict__ src, const int* __restrict__ dst,
                                const float* __restrict__ asrc, const float* __restrict__ adst,
                                const float* __restrict__ m, float* __restrict__ ssum) {
    int i = blockIdx.x * blockDim.x + threadIdx.x;
    if (i >= EE) return;
    int s = src[i], d = dst[i];
#pragma unroll
    for (int hd = 0; hd < HEADS; ++hd) {
        float e = leaky02(asrc[s * HEADS + hd] + adst[d * HEADS + hd]);
        atomicAdd(&ssum[d * HEADS + hd], expf(e - m[d * HEADS + hd]));
    }
}
__global__ void k_out_init(const float* __restrict__ hh, const float* __restrict__ asrc,
                           const float* __restrict__ adst, const float* __restrict__ m,
                           const float* __restrict__ ssum, float* __restrict__ outg) {
    int i = blockIdx.x * blockDim.x + threadIdx.x;
    if (i >= NN * HEADS * HID) return;
    int n = i >> 9, hc = i & 511, hd = hc >> 7;
    int nh = n * HEADS + hd;
    float alpha = expf(leaky02(asrc[nh] + adst[nh]) - m[nh]) / (ssum[nh] + 1e-16f);
    outg[i] = alpha * hh[i];
}
__global__ void k_out_scatter(const int* __restrict__ src, const int* __restrict__ dst,
                              const float* __restrict__ hh, const float* __restrict__ asrc,
                              const float* __restrict__ adst, const float* __restrict__ m,
                              const float* __restrict__ ssum, float* __restrict__ outg) {
    int e = blockIdx.x;
    int s = src[e], d = dst[e];
#pragma unroll
    for (int it = 0; it < 2; ++it) {
        int hc = it * 256 + threadIdx.x;
        int hd = hc >> 7;
        int dh = d * HEADS + hd;
        float ev = leaky02(asrc[s * HEADS + hd] + adst[dh]);
        float alpha = expf(ev - m[dh]) / (ssum[dh] + 1e-16f);
        atomicAdd(&outg[(size_t)d * (HEADS * HID) + hc],
                  alpha * hh[(size_t)s * (HEADS * HID) + hc]);
    }
}
__global__ void k_head_mean(const float* __restrict__ outg, const float* __restrict__ gb,
                            float* __restrict__ h) {
    int i = blockIdx.x * blockDim.x + threadIdx.x;
    if (i >= NN * HID) return;
    int n = i >> 7, c = i & 127;
    const float* o = outg + (size_t)n * (HEADS * HID);
    h[i] = 0.25f * (o[c] + o[HID + c] + o[2 * HID + c] + o[3 * HID + c]) + gb[c];
}

// ---------------- pooling ----------------
__global__ void k_pool_cnt(const int* __restrict__ batch, float* cnt) {
    int i = blockIdx.x * blockDim.x + threadIdx.x;
    if (i < NN) atomicAdd(&cnt[batch[i]], 1.0f);
}
__global__ void k_pool_sum(const int* __restrict__ batch, const float* __restrict__ h,
                           float* pooled) {
    int i = blockIdx.x * blockDim.x + threadIdx.x;
    if (i >= NN * HID) return;
    int n = i >> 7, c = i & 127;
    atomicAdd(&pooled[batch[n] * HID + c], h[i]);
}

// ---------------- MLP heads (128 -> 64 -> out_dim) ----------------
__global__ void k_mlp(const float* __restrict__ pooled, const float* __restrict__ cnt,
                      const float* __restrict__ W1, const float* __restrict__ b1,
                      const float* __restrict__ W2, const float* __restrict__ b2,
                      float* __restrict__ out, int out_dim, int out_base, int out_stride,
                      int do_softplus) {
    __shared__ float p[HID];
    __shared__ float z[64];
    int g = blockIdx.x;
    int tid = threadIdx.x;           // 64 threads
    float inv = 1.0f / fmaxf(cnt[g], 1.0f);
    p[tid]      = pooled[g * HID + tid] * inv;
    p[tid + 64] = pooled[g * HID + tid + 64] * inv;
    __syncthreads();
    float s = b1[tid];
    for (int k = 0; k < HID; ++k) s += p[k] * W1[k * 64 + tid];
    z[tid] = fmaxf(s, 0.f);
    __syncthreads();
    if (tid < out_dim) {
        float o = b2[tid];
        for (int j = 0; j < 64; ++j) o += z[j] * W2[j * out_dim + tid];
        if (do_softplus) o = (o > 20.f) ? o : log1pf(expf(o));
        out[out_base + g * out_stride + tid] = o;
    }
}

// =====================================================================
extern "C" void kernel_launch(void* const* d_in, const int* in_sizes, int n_in,
                              void* d_out, int out_size, void* d_ws, size_t ws_size,
                              hipStream_t stream) {
    // ---- inputs (depth-first insertion order of setup_inputs dict) ----
    const float* x      = (const float*)d_in[0];
    const int*   ei     = (const int*)d_in[1];
    const int*   batch  = (const int*)d_in[2];
    const float* embW   = (const float*)d_in[3];
    const float* embB   = (const float*)d_in[4];
    const float* convW[3] = {(const float*)d_in[5], (const float*)d_in[7], (const float*)d_in[9]};
    const float* convB[3] = {(const float*)d_in[6], (const float*)d_in[8], (const float*)d_in[10]};
    const float* bnG[3]   = {(const float*)d_in[11], (const float*)d_in[13], (const float*)d_in[15]};
    const float* bnB[3]   = {(const float*)d_in[12], (const float*)d_in[14], (const float*)d_in[16]};
    const float* gatW   = (const float*)d_in[17];   // [128 x 512]
    const float* attS   = (const float*)d_in[18];
    const float* attD   = (const float*)d_in[19];
    const float* gatB   = (const float*)d_in[20];
    const float* accW1  = (const float*)d_in[21]; const float* accB1 = (const float*)d_in[22];
    const float* accW2  = (const float*)d_in[23]; const float* accB2 = (const float*)d_in[24];
    const float* latW1  = (const float*)d_in[25]; const float* latB1 = (const float*)d_in[26];
    const float* latW2  = (const float*)d_in[27]; const float* latB2 = (const float*)d_in[28];
    const float* uncW1  = (const float*)d_in[29]; const float* uncB1 = (const float*)d_in[30];
    const float* uncW2  = (const float*)d_in[31]; const float* uncB2 = (const float*)d_in[32];

    const int* srcIdx = ei;
    const int* dstIdx = ei + EE;
    float* outp = (float*)d_out;

    // ---- workspace carve ----
    char* w = (char*)d_ws;
    size_t off = 0;
    auto alloc = [&](size_t bytes) -> void* {
        void* p = w + off;
        off = (off + bytes + 255) & ~(size_t)255;
        return p;
    };
    float* dinv   = (float*)alloc((size_t)NN * 4);
    float* h      = (float*)alloc((size_t)NN * HID * 4);
    float* t      = (float*)alloc((size_t)NN * HID * 4);
    float* agg    = (float*)alloc((size_t)NN * HID * 4);
    float* hh     = (float*)alloc((size_t)NN * HEADS * HID * 4);
    float* outg   = (float*)alloc((size_t)NN * HEADS * HID * 4);
    float* asrc   = (float*)alloc((size_t)NN * HEADS * 4);
    float* adst   = (float*)alloc((size_t)NN * HEADS * 4);
    float* mmax   = (float*)alloc((size_t)NN * HEADS * 4);
    float* ssum   = (float*)alloc((size_t)NN * HEADS * 4);
    float* stats  = (float*)alloc(256 * 4);          // musum | sqsum
    float* pooled = (float*)alloc((size_t)(GG * HID + GG) * 4);
    float* cnt    = pooled + GG * HID;
    __bf16* Bt    = (__bf16*)alloc((size_t)HEADS * HID * HID * 2);  // up to 512x128 bf16

    const int B = 256;
    auto blocks = [](long total, int bs) { return (int)((total + bs - 1) / bs); };

    // ---- degree + symmetric norm (self-loops folded in as +1) ----
    k_fill<<<blocks(NN, B), B, 0, stream>>>(dinv, 1.0f, NN);
    k_deg_scatter<<<blocks(EE, B), B, 0, stream>>>(dstIdx, dinv, EE);
    k_rsqrt_inplace<<<blocks(NN, B), B, 0, stream>>>(dinv, NN);

    // ---- embed ----
    k_embed<<<blocks((long)NN * HID, B), B, 0, stream>>>(x, embW, embB, h);

    // ---- 3 GCN layers ----
    for (int L = 0; L < 3; ++L) {
        k_wt_bf16<<<HID, HID, 0, stream>>>(convW[L], Bt, HID);
        k_gemm_bf16<<<NN / 16, 256, 0, stream>>>(h, Bt, t, HID, 8);
        k_conv_init<<<blocks((long)NN * HID, B), B, 0, stream>>>(t, dinv, convB[L], agg);
        k_conv_scatter<<<EE, HID, 0, stream>>>(srcIdx, dstIdx, t, dinv, agg);
        k_fill<<<1, 256, 0, stream>>>(stats, 0.0f, 256);
        k_bn_partial<<<160, HID, 0, stream>>>(agg, stats, stats + HID);
        k_bn_relu<<<blocks((long)NN * HID, B), B, 0, stream>>>(agg, stats, stats + HID,
                                                               bnG[L], bnB[L], h);
    }

    // ---- GAT ----
    k_wt_bf16<<<HEADS * HID, HID, 0, stream>>>(gatW, Bt, HEADS * HID);
    k_gemm_bf16<<<NN / 16, 256, 0, stream>>>(h, Bt, hh, HEADS * HID, 32);
    k_att_scores<<<blocks((long)NN * HEADS, B), B, 0, stream>>>(hh, attS, attD, asrc, adst);
    k_gat_m_init<<<blocks((long)NN * HEADS, B), B, 0, stream>>>(asrc, adst, mmax);
    k_gat_max<<<blocks(EE, B), B, 0, stream>>>(srcIdx, dstIdx, asrc, adst, mmax);
    k_gat_s_init<<<blocks((long)NN * HEADS, B), B, 0, stream>>>(asrc, adst, mmax, ssum);
    k_gat_s_scatter<<<blocks(EE, B), B, 0, stream>>>(srcIdx, dstIdx, asrc, adst, mmax, ssum);
    k_out_init<<<blocks((long)NN * HEADS * HID, B), B, 0, stream>>>(hh, asrc, adst, mmax, ssum, outg);
    k_out_scatter<<<EE, 256, 0, stream>>>(srcIdx, dstIdx, hh, asrc, adst, mmax, ssum, outg);
    k_head_mean<<<blocks((long)NN * HID, B), B, 0, stream>>>(outg, gatB, h);

    // ---- global mean pool ----
    k_fill<<<blocks(GG * HID + GG, B), B, 0, stream>>>(pooled, 0.0f, GG * HID + GG);
    k_pool_cnt<<<blocks(NN, B), B, 0, stream>>>(batch, cnt);
    k_pool_sum<<<blocks((long)NN * HID, B), B, 0, stream>>>(batch, h, pooled);

    // ---- MLP heads: preds = [acc | lat] (G x 2), then unc (G x 2, softplus) ----
    k_mlp<<<GG, 64, 0, stream>>>(pooled, cnt, accW1, accB1, accW2, accB2, outp, 1, 0,       2, 0);
    k_mlp<<<GG, 64, 0, stream>>>(pooled, cnt, latW1, latB1, latW2, latB2, outp, 1, 1,       2, 0);
    k_mlp<<<GG, 64, 0, stream>>>(pooled, cnt, uncW1, uncB1, uncW2, uncB2, outp, 2, GG * 2,  2, 1);
}
